// Attention_88510686036759
// MI455X (gfx1250) — matrix operand order repaired and checked by reading, not compiled, not run on hardware
//
#include <hip/hip_runtime.h>
#include <hip/hip_bf16.h>
#include <math.h>

typedef __bf16 bf16;
typedef __attribute__((ext_vector_type(16))) bf16 v16bf;
typedef __attribute__((ext_vector_type(8)))  float v8f;

#define BB 16
#define CC 2048
#define QQ 128
#define EE 200
#define EP 224          // E padded to 7*32
#define FF 896          // 4*EP projection K
#define OO 128

#define TILE_ELEMS 512  // 32 lanes * 16 bf16 per swizzled fragment tile

// ---------------- fragment load: one 32B vector load per lane ----------------
__device__ __forceinline__ v16bf frag(const bf16* tile) {
  return *(const v16bf*)(tile + (threadIdx.x & 31) * 16);
}

__device__ __forceinline__ v8f wmma_bf16(v16bf a, v16bf b, v8f c) {
  return __builtin_amdgcn_wmma_f32_16x16x32_bf16(false, a, false, b, (short)0, c,
                                                 false, false);
}

// A-layout forward: VGPR r, pair bit bb -> k within 32
__device__ __forceinline__ int a_k(int r, int half, int bb) {
  return half * 16 + r * 2 + bb;
}
// A-layout inverse: k within 32 -> j (element index in v16bf); half = (k>>4)&1
__device__ __forceinline__ int a_j(int kin) {
  return kin & 15;
}

// ---------------- ordered-uint float atomic max ----------------
__device__ __forceinline__ unsigned f2ord(float f) {
  unsigned u = __float_as_uint(f);
  return (u & 0x80000000u) ? ~u : (u | 0x80000000u);
}
__device__ __forceinline__ float ord2f(unsigned u) {
  return (u & 0x80000000u) ? __uint_as_float(u & 0x7FFFFFFFu)
                           : __uint_as_float(~u);
}

__device__ __forceinline__ float rmax16(float v) {
#pragma unroll
  for (int m = 1; m < 16; m <<= 1) v = fmaxf(v, __shfl_xor(v, m, 32));
  return v;
}
__device__ __forceinline__ float rsum16(float v) {
#pragma unroll
  for (int m = 1; m < 16; m <<= 1) v += __shfl_xor(v, m, 32);
  return v;
}
__device__ __forceinline__ float rsum32(float v) {
#pragma unroll
  for (int m = 1; m < 32; m <<= 1) v += __shfl_xor(v, m, 32);
  return v;
}

// ================= init =================
__global__ void k_init(unsigned* colmax_u, float* colsum) {
  int i = blockIdx.x * blockDim.x + threadIdx.x;
  if (i < BB * QQ) {
    colmax_u[i] = f2ord(-INFINITY);
    colsum[i] = 0.f;
  }
}

// ======== row dot products (wave per row) ========
__global__ void k_sc(const float* xc, const float* w_sim, float* s_c) {
  int wave = threadIdx.x >> 5, lane = threadIdx.x & 31;
  int row = blockIdx.x * 8 + wave;
  int b = row >> 11, c = row & (CC - 1);
  const float* src = xc + ((size_t)b * CC + c) * EE;
  float dot = 0.f;
#pragma unroll 1
  for (int e = lane; e < EE; e += 32) dot += src[e] * w_sim[e];
  dot = rsum32(dot);
  if (lane == 0) s_c[b * CC + c] = dot;
}

__global__ void k_sq(const float* xq, const float* w_sim, float* s_q) {
  int wave = threadIdx.x >> 5, lane = threadIdx.x & 31;
  int row = blockIdx.x * 8 + wave;
  int b = row >> 7, q = row & (QQ - 1);
  const float* src = xq + ((size_t)b * QQ + q) * EE;
  float dot = 0.f;
#pragma unroll 1
  for (int e = lane; e < EE; e += 32) dot += src[e] * w_sim[EE + e];
  dot = rsum32(dot);
  if (lane == 0) s_q[b * QQ + q] = dot;
}

// ======== xcw3 in swizzled A layout (M=CC, K=EP), wave per tile ========
__global__ void k_xcw3_A(const float* xc, const float* w_sim, bf16* dst) {
  int wave = threadIdx.x >> 5, lane = threadIdx.x & 31;
  int t = blockIdx.x * 8 + wave;            // total BB*128*7
  int kt = t % (EP / 32);
  int rest = t / (EP / 32);
  int mtile = rest % (CC / 16);
  int b = rest / (CC / 16);
  int m = lane & 15, half = lane >> 4;
  const float* src = xc + ((size_t)b * CC + mtile * 16 + m) * EE;
  v16bf v;
#pragma unroll
  for (int r = 0; r < 8; ++r)
#pragma unroll
    for (int bb = 0; bb < 2; ++bb) {
      int k = kt * 32 + a_k(r, half, bb);
      float x = (k < EE) ? src[k] * w_sim[2 * EE + k] : 0.f;
      v[2 * r + bb] = (bf16)x;
    }
  *(v16bf*)(dst + (size_t)t * TILE_ELEMS + lane * 16) = v;
}

// ======== xc in swizzled B layout (K=CC, N=EP), wave per tile ========
__global__ void k_xc_B(const float* xc, bf16* dst) {
  int wave = threadIdx.x >> 5, lane = threadIdx.x & 31;
  int t = blockIdx.x * 8 + wave;            // total BB*14*64
  int kt = t % (CC / 32);
  int rest = t / (CC / 32);
  int ntile = rest % (EP / 16);
  int b = rest / (EP / 16);
  int n = lane & 15, half = lane >> 4;
  int e = ntile * 16 + n;
  v16bf v;
#pragma unroll
  for (int r = 0; r < 8; ++r)
#pragma unroll
    for (int bb = 0; bb < 2; ++bb) {
      int c = kt * 32 + half * 16 + 2 * r + bb;
      float x = (e < EE) ? xc[((size_t)b * CC + c) * EE + e] : 0.f;
      v[2 * r + bb] = (bf16)x;
    }
  *(v16bf*)(dst + (size_t)t * TILE_ELEMS + lane * 16) = v;
}

// ======== xq^T in swizzled B layout (K=EP, N=QQ) ========
__global__ void k_xqT_B(const float* xq, bf16* dst) {
  int wave = threadIdx.x >> 5, lane = threadIdx.x & 31;
  int t = blockIdx.x * 8 + wave;            // total BB*8*7
  int kt = t % (EP / 32);
  int rest = t / (EP / 32);
  int ntile = rest % (QQ / 16);
  int b = rest / (QQ / 16);
  int n = lane & 15, half = lane >> 4;
  const float* src = xq + ((size_t)b * QQ + ntile * 16 + n) * EE;
  v16bf v;
#pragma unroll
  for (int r = 0; r < 8; ++r)
#pragma unroll
    for (int bb = 0; bb < 2; ++bb) {
      int e = kt * 32 + half * 16 + 2 * r + bb;
      v[2 * r + bb] = (bf16)((e < EE) ? src[e] : 0.f);
    }
  *(v16bf*)(dst + (size_t)t * TILE_ELEMS + lane * 16) = v;
}

// ======== xq in swizzled B layout (K=QQ, N=EP) ========
__global__ void k_xq_B(const float* xq, bf16* dst) {
  int wave = threadIdx.x >> 5, lane = threadIdx.x & 31;
  int t = blockIdx.x * 8 + wave;            // total BB*14*4
  int kt = t % (QQ / 32);
  int rest = t / (QQ / 32);
  int ntile = rest % (EP / 16);
  int b = rest / (EP / 16);
  int n = lane & 15, half = lane >> 4;
  int e = ntile * 16 + n;
  v16bf v;
#pragma unroll
  for (int r = 0; r < 8; ++r)
#pragma unroll
    for (int bb = 0; bb < 2; ++bb) {
      int q = kt * 32 + half * 16 + 2 * r + bb;
      float x = (e < EE) ? xq[((size_t)b * QQ + q) * EE + e] : 0.f;
      v[2 * r + bb] = (bf16)x;
    }
  *(v16bf*)(dst + (size_t)t * TILE_ELEMS + lane * 16) = v;
}

// ======== w_proj^T in swizzled B layout (K=FF, N=OO) ========
__global__ void k_wproj_B(const float* w_proj, bf16* dst) {
  int wave = threadIdx.x >> 5, lane = threadIdx.x & 31;
  int t = blockIdx.x * 8 + wave;            // total 8*28
  if (t >= (OO / 16) * (FF / 32)) return;
  int kt = t % (FF / 32);
  int ntile = t / (FF / 32);
  int n = lane & 15, half = lane >> 4;
  int o = ntile * 16 + n;
  v16bf v;
#pragma unroll
  for (int r = 0; r < 8; ++r)
#pragma unroll
    for (int bb = 0; bb < 2; ++bb) {
      int f = kt * 32 + half * 16 + 2 * r + bb;
      int g = f / EP, e = f % EP;
      float x = (e < EE) ? w_proj[o * (4 * EE) + g * EE + e] : 0.f;
      v[2 * r + bb] = (bf16)x;
    }
  *(v16bf*)(dst + (size_t)t * TILE_ELEMS + lane * 16) = v;
}

// ======= scores GEMM + row stats + col max (1 wave per 16 rows) =======
__global__ void k_scores(const bf16* xcw3, const bf16* xqT, const float* s_c,
                         const float* s_q, float* scores, float* rowmax,
                         float* rowsum, unsigned* colmax_u) {
  const int b = blockIdx.y;
  const int mtile = blockIdx.x;
  const int c0 = mtile * 16;
  const int lane = threadIdx.x & 31;
  const int lh = lane & 15, half = lane >> 4;

  const bf16* At = xcw3 + (((size_t)b * (CC / 16) + mtile) * (EP / 32)) * TILE_ELEMS;
  const bf16* Bt = xqT + ((size_t)b * (QQ / 16) * (EP / 32)) * TILE_ELEMS;

  v8f acc[8];
#pragma unroll
  for (int n = 0; n < 8; ++n) acc[n] = (v8f){0, 0, 0, 0, 0, 0, 0, 0};

  // fully unrolled, loads grouped per k-step so WMMAs overlap in-flight loads
#pragma unroll
  for (int kt = 0; kt < EP / 32; ++kt) {
    v16bf a = frag(At + (size_t)kt * TILE_ELEMS);
    v16bf bfr[8];
#pragma unroll
    for (int n = 0; n < 8; ++n)
      bfr[n] = frag(Bt + ((size_t)n * (EP / 32) + kt) * TILE_ELEMS);
#pragma unroll
    for (int n = 0; n < 8; ++n) acc[n] = wmma_bf16(a, bfr[n], acc[n]);
  }

  float scv[8];
#pragma unroll
  for (int r = 0; r < 8; ++r) scv[r] = s_c[b * CC + c0 + r + 8 * half];
#pragma unroll
  for (int n = 0; n < 8; ++n) {
    float sqv = s_q[b * QQ + n * 16 + lh];
#pragma unroll
    for (int r = 0; r < 8; ++r) acc[n][r] += scv[r] + sqv;
  }

#pragma unroll
  for (int r = 0; r < 8; ++r) {
    int c = c0 + r + 8 * half;
    float vmax = -INFINITY;
#pragma unroll
    for (int n = 0; n < 8; ++n) vmax = fmaxf(vmax, acc[n][r]);
    vmax = rmax16(vmax);
    float vsum = 0.f;
#pragma unroll
    for (int n = 0; n < 8; ++n) {
      scores[((size_t)b * CC + c) * QQ + n * 16 + lh] = acc[n][r];
      vsum += __expf(acc[n][r] - vmax);
    }
    vsum = rsum16(vsum);
    if (lh == 0) {
      rowmax[b * CC + c] = vmax;
      rowsum[b * CC + c] = vsum;
    }
  }

#pragma unroll
  for (int n = 0; n < 8; ++n) {
    float cm = -INFINITY;
#pragma unroll
    for (int r = 0; r < 8; ++r) cm = fmaxf(cm, acc[n][r]);
    cm = fmaxf(cm, __shfl_xor(cm, 16, 32));
    if (half == 0) atomicMax(&colmax_u[b * QQ + n * 16 + lh], f2ord(cm));
  }
}

// ================= column sum of exp =================
__global__ void k_colsum(const float* scores, const unsigned* colmax_u,
                         float* colsum) {
  int b = blockIdx.y;
  int q = threadIdx.x;
  int c0 = blockIdx.x * 128;
  float cm = ord2f(colmax_u[b * QQ + q]);
  float s = 0.f;
#pragma unroll 1
  for (int c = c0; c < c0 + 128; ++c)
    s += __expf(scores[((size_t)b * CC + c) * QQ + q] - cm);
  atomicAdd(&colsum[b * QQ + q], s);
}

// ======== S1 in swizzled A layout (M=CC, K=QQ), wave per tile ========
__global__ void k_s1_A(const float* scores, const float* rowmax,
                       const float* rowsum, bf16* dst) {
  int wave = threadIdx.x >> 5, lane = threadIdx.x & 31;
  int t = blockIdx.x * 8 + wave;            // total BB*128*4
  int kt = t % (QQ / 32);
  int rest = t / (QQ / 32);
  int mtile = rest % (CC / 16);
  int b = rest / (CC / 16);
  int m = lane & 15, half = lane >> 4;
  int c = mtile * 16 + m;
  float rm = rowmax[b * CC + c];
  float rinv = 1.f / rowsum[b * CC + c];
  const float* srow = scores + ((size_t)b * CC + c) * QQ;
  v16bf v;
#pragma unroll
  for (int r = 0; r < 8; ++r)
#pragma unroll
    for (int bb = 0; bb < 2; ++bb) {
      int k = kt * 32 + a_k(r, half, bb);
      v[2 * r + bb] = (bf16)(__expf(srow[k] - rm) * rinv);
    }
  *(v16bf*)(dst + (size_t)t * TILE_ELEMS + lane * 16) = v;
}

// ======== S2^T in swizzled A layout (M=QQ, K=CC), wave per tile ========
__global__ void k_s2t_A(const float* scores, const unsigned* colmax_u,
                        const float* colsum, bf16* dst) {
  int wave = threadIdx.x >> 5, lane = threadIdx.x & 31;
  int t = blockIdx.x * 8 + wave;            // total BB*8*64
  int kt = t % (CC / 32);
  int rest = t / (CC / 32);
  int mtile = rest % (QQ / 16);
  int b = rest / (QQ / 16);
  int m = lane & 15, half = lane >> 4;
  int q = mtile * 16 + m;
  float cm = ord2f(colmax_u[b * QQ + q]);
  float inv = 1.f / colsum[b * QQ + q];
  v16bf v;
#pragma unroll
  for (int r = 0; r < 8; ++r)
#pragma unroll
    for (int bb = 0; bb < 2; ++bb) {
      int c = kt * 32 + a_k(r, half, bb);
      float s = scores[((size_t)b * CC + c) * QQ + q];
      v[2 * r + bb] = (bf16)(__expf(s - cm) * inv);
    }
  *(v16bf*)(dst + (size_t)t * TILE_ELEMS + lane * 16) = v;
}

// ======= Aq = S2^T (QxC) @ xc (CxEP); output in swizzled B layout =======
__global__ void k_aq(const bf16* s2t, const bf16* xc_swz, bf16* aq) {
  const int b = blockIdx.z;
  const int mtile = blockIdx.y;             // q-row tile (0..7)
  const int nch = blockIdx.x;               // 0..1, 7 n-tiles each
  const int lane = threadIdx.x & 31;
  const int lh = lane & 15, half = lane >> 4;

  const bf16* At = s2t + (((size_t)b * (QQ / 16) + mtile) * (CC / 32)) * TILE_ELEMS;
  const bf16* Bt = xc_swz + ((size_t)b * (EP / 16) * (CC / 32)) * TILE_ELEMS;

  v8f acc[7];
#pragma unroll
  for (int n = 0; n < 7; ++n) acc[n] = (v8f){0, 0, 0, 0, 0, 0, 0, 0};

#pragma unroll 2
  for (int kt = 0; kt < CC / 32; ++kt) {
    // group all fragment loads for this k-step before the WMMA chain
    v16bf a = frag(At + (size_t)kt * TILE_ELEMS);
    v16bf bfr[7];
#pragma unroll
    for (int n = 0; n < 7; ++n)
      bfr[n] = frag(Bt + (((size_t)(nch * 7 + n)) * (CC / 32) + kt) * TILE_ELEMS);
    __builtin_prefetch(At + (size_t)(kt + 1) * TILE_ELEMS, 0, 0);
#pragma unroll
    for (int n = 0; n < 7; ++n) acc[n] = wmma_bf16(a, bfr[n], acc[n]);
  }

  // store into swizzled B layout for aq (K=QQ, N=EP)
#pragma unroll
  for (int n = 0; n < 7; ++n) {
    int ntile = nch * 7 + n;
#pragma unroll
    for (int r = 0; r < 8; ++r) {
      int qrow = mtile * 16 + r + 8 * half;  // K index
      int kt = qrow >> 5, kin = qrow & 31;
      int lane_d = ((kin >> 4) & 1) * 16 + lh;
      int j = kin & 15;
      size_t idx = ((((size_t)b * (EP / 16) + ntile) * (QQ / 32) + kt) * 32 + lane_d) * 16 + j;
      aq[idx] = (bf16)acc[n][r];
    }
  }
}

// ===== fused: c2q + q2c WMMA, feats -> LDS (A-swizzled), projection WMMA =====
__global__ void k_main(const bf16* s1_swz, const bf16* xq_swz, const bf16* aq_swz,
                       const float* xc, const bf16* wpT, const float* b_proj,
                       float* out) {
  __shared__ bf16 sf[(FF / 32) * TILE_ELEMS];  // 28 KB, A-swizzled feats
  const int b = blockIdx.y;
  const int mtile = blockIdx.x;
  const int c0 = mtile * 16;
  const int lane = threadIdx.x & 31;
  const int lh = lane & 15, half = lane >> 4;

  // S1 A-fragments: 4 contiguous swizzled tiles
  const bf16* S1t = s1_swz + (((size_t)b * (CC / 16) + mtile) * (QQ / 32)) * TILE_ELEMS;
  v16bf afr[4];
#pragma unroll
  for (int kt = 0; kt < 4; ++kt) afr[kt] = frag(S1t + (size_t)kt * TILE_ELEMS);

  const bf16* Bq = xq_swz + ((size_t)b * (EP / 16) * (QQ / 32)) * TILE_ELEMS;
  const bf16* Ba = aq_swz + ((size_t)b * (EP / 16) * (QQ / 32)) * TILE_ELEMS;

#pragma unroll 1
  for (int nt = 0; nt < EP / 16; ++nt) {
    // all 8 B fragments for this n-tile loaded before the WMMA chain
    v16bf bq[4], ba[4];
#pragma unroll
    for (int kt = 0; kt < 4; ++kt) {
      bq[kt] = frag(Bq + ((size_t)nt * (QQ / 32) + kt) * TILE_ELEMS);
      ba[kt] = frag(Ba + ((size_t)nt * (QQ / 32) + kt) * TILE_ELEMS);
    }
    v8f ac = (v8f){0, 0, 0, 0, 0, 0, 0, 0};   // c2q
    v8f aq2 = (v8f){0, 0, 0, 0, 0, 0, 0, 0};  // q2c
#pragma unroll
    for (int kt = 0; kt < 4; ++kt) {
      ac = wmma_bf16(afr[kt], bq[kt], ac);
      aq2 = wmma_bf16(afr[kt], ba[kt], aq2);
    }
    int e = nt * 16 + lh;
#pragma unroll
    for (int r = 0; r < 8; ++r) {
      int rowl = r + 8 * half;               // m within tile
      float x = (e < EE) ? xc[((size_t)b * CC + c0 + rowl) * EE + e] : 0.f;
      float fv[4] = {x, ac[r], x * ac[r], x * aq2[r]};
#pragma unroll
      for (int g = 0; g < 4; ++g) {
        int f = g * EP + e;
        int ktf = f >> 5, kin = f & 31;
        int hf = (kin >> 4) & 1;
        sf[((ktf * 32) + hf * 16 + rowl) * 16 + a_j(kin)] = (bf16)fv[g];
      }
    }
  }
  __syncthreads();

  // projection: feats (16xFF) @ wpT (FFxOO)
  v8f acc[8];
#pragma unroll
  for (int n = 0; n < 8; ++n) acc[n] = (v8f){0, 0, 0, 0, 0, 0, 0, 0};

#pragma unroll 1
  for (int kt = 0; kt < FF / 32; ++kt) {
    v16bf a = frag(&sf[kt * TILE_ELEMS]);    // ds_load_b128 x2
    v16bf bw[8];
#pragma unroll
    for (int n = 0; n < 8; ++n)
      bw[n] = frag(wpT + ((size_t)n * (FF / 32) + kt) * TILE_ELEMS);
    __builtin_prefetch(wpT + ((size_t)kt + 1) * TILE_ELEMS, 0, 0);
#pragma unroll
    for (int n = 0; n < 8; ++n) acc[n] = wmma_bf16(a, bw[n], acc[n]);
  }

#pragma unroll
  for (int n = 0; n < 8; ++n) {
    float bias = b_proj[n * 16 + lh];
#pragma unroll
    for (int r = 0; r < 8; ++r) {
      int c = c0 + r + 8 * half;
      out[((size_t)b * CC + c) * OO + n * 16 + lh] = acc[n][r] + bias;
    }
  }
}

// ============================ launcher ============================
extern "C" void kernel_launch(void* const* d_in, const int* in_sizes, int n_in,
                              void* d_out, int out_size, void* d_ws,
                              size_t ws_size, hipStream_t stream) {
  const float* xc = (const float*)d_in[0];
  const float* xq = (const float*)d_in[1];
  const float* w_sim = (const float*)d_in[2];
  const float* w_proj = (const float*)d_in[3];
  const float* b_proj = (const float*)d_in[4];
  float* out = (float*)d_out;

  char* w = (char*)d_ws;
  size_t off = 0;
  auto alloc = [&](size_t bytes) -> void* {
    void* p = w + off;
    off = (off + bytes + 255) & ~(size_t)255;
    return p;
  };

  const size_t TB = TILE_ELEMS * sizeof(bf16);  // 1 KB per tile
  bf16* xcw3_swz = (bf16*)alloc((size_t)BB * (CC / 16) * (EP / 32) * TB);
  bf16* xc_swz   = (bf16*)alloc((size_t)BB * (EP / 16) * (CC / 32) * TB);
  bf16* xqT_swz  = (bf16*)alloc((size_t)BB * (QQ / 16) * (EP / 32) * TB);
  bf16* xq_swz   = (bf16*)alloc((size_t)BB * (EP / 16) * (QQ / 32) * TB);
  bf16* wpT_swz  = (bf16*)alloc((size_t)(OO / 16) * (FF / 32) * TB);
  bf16* s1_swz   = (bf16*)alloc((size_t)BB * (CC / 16) * (QQ / 32) * TB);
  bf16* s2t_swz  = (bf16*)alloc((size_t)BB * (QQ / 16) * (CC / 32) * TB);
  bf16* aq_swz   = (bf16*)alloc((size_t)BB * (EP / 16) * (QQ / 32) * TB);
  float* s_c     = (float*)alloc((size_t)BB * CC * 4);
  float* s_q     = (float*)alloc((size_t)BB * QQ * 4);
  float* scores  = (float*)alloc((size_t)BB * CC * QQ * 4);
  float* rowmax  = (float*)alloc((size_t)BB * CC * 4);
  float* rowsum  = (float*)alloc((size_t)BB * CC * 4);
  unsigned* colmax_u = (unsigned*)alloc((size_t)BB * QQ * 4);
  float* colsum  = (float*)alloc((size_t)BB * QQ * 4);

  k_init<<<8, 256, 0, stream>>>(colmax_u, colsum);
  k_sc<<<BB * CC / 8, 256, 0, stream>>>(xc, w_sim, s_c);
  k_sq<<<BB * QQ / 8, 256, 0, stream>>>(xq, w_sim, s_q);
  k_xcw3_A<<<BB * (CC / 16) * (EP / 32) / 8, 256, 0, stream>>>(xc, w_sim, xcw3_swz);
  k_xc_B<<<BB * (EP / 16) * (CC / 32) / 8, 256, 0, stream>>>(xc, xc_swz);
  k_xqT_B<<<BB * (QQ / 16) * (EP / 32) / 8, 256, 0, stream>>>(xq, xqT_swz);
  k_xq_B<<<BB * (EP / 16) * (QQ / 32) / 8, 256, 0, stream>>>(xq, xq_swz);
  k_wproj_B<<<28, 256, 0, stream>>>(w_proj, wpT_swz);

  k_scores<<<dim3(CC / 16, BB), 32, 0, stream>>>(xcw3_swz, xqT_swz, s_c, s_q,
                                                 scores, rowmax, rowsum, colmax_u);
  k_colsum<<<dim3(CC / 128, BB), 128, 0, stream>>>(scores, colmax_u, colsum);
  k_s1_A<<<BB * (CC / 16) * (QQ / 32) / 8, 256, 0, stream>>>(scores, rowmax,
                                                             rowsum, s1_swz);
  k_s2t_A<<<BB * (QQ / 16) * (CC / 32) / 8, 256, 0, stream>>>(scores, colmax_u,
                                                              colsum, s2t_swz);
  k_aq<<<dim3(2, QQ / 16, BB), 32, 0, stream>>>(s2t_swz, xc_swz, aq_swz);
  k_main<<<dim3(CC / 16, BB), 32, 0, stream>>>(s1_swz, xq_swz, aq_swz, xc,
                                               wpT_swz, b_proj, out);
}
